// BlockLoRA_30906584662342
// MI455X (gfx1250) — matrix operand order repaired and checked by reading, not compiled, not run on hardware
//
#include <hip/hip_runtime.h>
#include <hip/hip_bf16.h>

// ---------------- problem constants ----------------
#define C_DIM   384
#define HQn     8
#define HKVn    2
#define HDn     48
#define HDP     64          // padded head dim for WMMA (K multiple of 32)
#define R_LORA  4
#define E_NUM   4
#define T_SEQ   2048
#define B_NUM   4
#define NROWS   (B_NUM*T_SEQ)   // 8192
#define CAPS    2560            // ceil(1.25*8192/4)
#define FFN     1536
#define SCALE_LORA 0.25f

typedef __attribute__((ext_vector_type(16))) __bf16 bf16x16;
typedef __attribute__((ext_vector_type(8)))  float  v8f;
typedef __attribute__((ext_vector_type(4)))  int    v4i;

union FragA { bf16x16 v; unsigned int u[8]; };

#define WMMA_BF16(Afr, Bfr, Cacc) \
  __builtin_amdgcn_wmma_f32_16x16x32_bf16(false, (Afr).v, false, (Bfr).v, (short)0, (Cacc), false, false)

// CDNA5 async global->LDS copy path (ASYNCcnt), guarded for toolchain support
#if defined(__gfx1250__) && __has_builtin(__builtin_amdgcn_global_load_async_to_lds_b128) && __has_builtin(__builtin_amdgcn_s_wait_asynccnt)
#define HAVE_ASYNC 1
typedef __attribute__((address_space(1))) v4i GV4;
typedef __attribute__((address_space(3))) v4i LV4;
__device__ __forceinline__ void async_b128(const void* g, void* l) {
    __builtin_amdgcn_global_load_async_to_lds_b128((GV4*)g, (LV4*)l, 0, 0);
}
__device__ __forceinline__ void async_wait0() { __builtin_amdgcn_s_wait_asynccnt(0); }
#else
#define HAVE_ASYNC 0
#endif

__device__ __forceinline__ unsigned short f2bf(float f) {
    unsigned int u = __float_as_uint(f);
    u += 0x7fffu + ((u >> 16) & 1u);       // RNE
    return (unsigned short)(u >> 16);
}

// ---------------- zero init ----------------
__global__ void zero_u32(unsigned int* p, long n) {
    long i = (long)blockIdx.x * blockDim.x + threadIdx.x;
    long st = (long)gridDim.x * blockDim.x;
    for (; i < n; i += st) p[i] = 0u;
}

// ------- fold LoRA into weights, convert bf16, store TRANSPOSED: out[n, k] -------
// out[(colOff+o)*Kin + c] = W[c,o] + SCALE * sum_r A[r,c] * B[o,r]
__global__ void prep_weight(const float* __restrict__ W, long sW,
                            const float* __restrict__ A, long sA, long aAid,
                            const float* __restrict__ Bm, long sB, long bAid,
                            unsigned short* __restrict__ out, long sO,
                            int Kin, int Nout, int colOff,
                            const int* aidp)
{
    int e = blockIdx.z;
    int aid = aidp[0];
    const float* Wp = W  + (long)e*sW;
    const float* Ap = A  + (long)e*sA + (long)aid*aAid;
    const float* Bp = Bm + (long)e*sB + (long)aid*bAid;
    unsigned short* op = out + (long)e*sO;
    long total = (long)Kin * Nout;
    long i  = (long)blockIdx.x * blockDim.x + threadIdx.x;
    long st = (long)gridDim.x * blockDim.x;
    for (; i < total; i += st) {
        int c = (int)(i / Nout);
        int o = (int)(i % Nout);
        float v = Wp[(long)c*Nout + o];
        #pragma unroll
        for (int r = 0; r < R_LORA; r++)
            v += SCALE_LORA * Ap[r*Kin + c] * Bp[o*R_LORA + r];
        op[(long)(colOff + o)*Kin + c] = f2bf(v);
    }
}

// ---------------- LayerNorm (one 128-thread block per row) ----------------
__global__ void ln_kernel(const float* __restrict__ x, const float* __restrict__ g,
                          const float* __restrict__ b, float* outf, unsigned short* outb)
{
    int row = blockIdx.x, tid = threadIdx.x;
    const float* xr = x + (long)row * C_DIM;
    float v0 = xr[tid], v1 = xr[tid+128], v2 = xr[tid+256];
    float s = v0+v1+v2, s2 = v0*v0+v1*v1+v2*v2;
    #pragma unroll
    for (int off = 16; off; off >>= 1) { s += __shfl_xor(s, off, 32); s2 += __shfl_xor(s2, off, 32); }
    __shared__ float red[8];
    int wid = tid >> 5, lane = tid & 31;
    if (lane == 0) { red[wid] = s; red[4+wid] = s2; }
    __syncthreads();
    float sum  = red[0]+red[1]+red[2]+red[3];
    float sum2 = red[4]+red[5]+red[6]+red[7];
    float mean = sum  * (1.0f/C_DIM);
    float var  = sum2 * (1.0f/C_DIM) - mean*mean;
    float rstd = rsqrtf(var + 1e-5f);
    float vals[3] = {v0, v1, v2};
    #pragma unroll
    for (int i = 0; i < 3; i++) {
        int c = tid + 128*i;
        float y = (vals[i] - mean)*rstd*g[c] + b[c];
        if (outf) outf[(long)row*C_DIM + c] = y;
        outb[(long)row*C_DIM + c] = f2bf(y);
    }
}

// ------------- generic bf16 WMMA GEMM: C[M,N] = A[M,K] @ Bt[N,K]^T -------------
// block 256 = 8 waves; block tile 256x64; wave tile 32x64 (8 WMMA / K-step)
// mode 0: Cf = acc ; mode 1: Cf = acc + Res ; mode 2: Cb = bf16(relu(acc))
__global__ __launch_bounds__(256) void gemm_bf16(
        const unsigned short* __restrict__ A, long sA,
        const unsigned short* __restrict__ Bt, long sB,   // transposed weights [N,K]
        float* Cf, unsigned short* Cb, long sC,
        const float* __restrict__ Res,
        int M, int N, int K, int mode)
{
    __shared__ unsigned int ldsA[4096];   // 256 rows x 16 dwords (32 bf16 K-slice)
    __shared__ unsigned int ldsB[1024];   // 64 cols  x 16 dwords (K contiguous)
    int e = blockIdx.z;
    const unsigned short* Ag  = A  + (long)e*sA;
    const unsigned short* Btg = Bt + (long)e*sB;
    const unsigned int* Agu  = (const unsigned int*)Ag;
    const unsigned int* Btgu = (const unsigned int*)Btg;
    int tid = threadIdx.x, lane = tid & 31, w = tid >> 5;
    int m0 = blockIdx.y * 256, n0 = blockIdx.x * 64;
    int Kd = K >> 1;
    int koff2 = (lane < 16) ? 0 : 4;      // A-frag: K lo/hi half per lane-half
    int jb    = (lane < 16) ? 0 : 8;      // B-frag: same K order as A-frag (koff2)
    int nlane = lane & 15;
    v8f acc[8];
    #pragma unroll
    for (int i = 0; i < 8; i++) acc[i] = (v8f){};

    for (int k0 = 0; k0 < K; k0 += 32) {
        if (k0 + 32 < K)
            __builtin_prefetch(&Agu[(long)(m0 + tid)*Kd + ((k0+32) >> 1)], 0, 1);
#if HAVE_ASYNC
        #pragma unroll
        for (int i = 0; i < 4; i++) {     // A: 256 rows x 64B -> 1024 b128, 4/thread
            int idx = tid + i*256;
            int r = idx >> 2, seg = idx & 3;
            async_b128(Ag + (long)(m0 + r)*K + k0 + seg*8,
                       (char*)ldsA + r*64 + seg*16);
        }
        {                                  // B: 64 rows x 64B -> 256 b128, 1/thread
            int n = tid >> 2, seg = tid & 3;
            async_b128(Btg + (long)(n0 + n)*K + k0 + seg*8,
                       (char*)ldsB + n*64 + seg*16);
        }
        async_wait0();
#else
        for (int s = tid; s < 4096; s += 256)
            ldsA[s] = Agu[(long)(m0 + (s >> 4))*Kd + (k0 >> 1) + (s & 15)];
        for (int s = tid; s < 1024; s += 256)
            ldsB[s] = Btgu[(long)(n0 + (s >> 4))*Kd + (k0 >> 1) + (s & 15)];
#endif
        __syncthreads();
        FragA a0, a1;
        {
            const unsigned int* ar0 = &ldsA[((w << 5) + nlane) * 16];
            const unsigned int* ar1 = ar0 + 16*16;
            #pragma unroll
            for (int j = 0; j < 4; j++) {
                a0.u[j] = ar0[koff2+j]; a0.u[4+j] = ar0[8+koff2+j];
                a1.u[j] = ar1[koff2+j]; a1.u[4+j] = ar1[8+koff2+j];
            }
        }
        #pragma unroll
        for (int nt = 0; nt < 4; nt++) {
            FragA b;
            #pragma unroll
            for (int j = 0; j < 8; j++) b.u[j] = ldsB[(nt*16 + nlane)*16 + koff2 + (j & 3) + ((j >> 2) << 3)];
            acc[nt]     = WMMA_BF16(a0, b, acc[nt]);
            acc[4 + nt] = WMMA_BF16(a1, b, acc[4 + nt]);
        }
        __syncthreads();
    }
    int halfo = (lane < 16) ? 0 : 8;
    int rbase = m0 + (w << 5) + halfo;
    if (mode == 0) {
        #pragma unroll
        for (int sub = 0; sub < 2; sub++)
        #pragma unroll
        for (int nt = 0; nt < 4; nt++)
        #pragma unroll
        for (int r = 0; r < 8; r++) {
            int row = rbase + sub*16 + r, col = n0 + nt*16 + nlane;
            Cf[(long)e*sC + (long)row*N + col] = acc[sub*4+nt][r];
        }
    } else if (mode == 1) {
        #pragma unroll
        for (int sub = 0; sub < 2; sub++)
        #pragma unroll
        for (int nt = 0; nt < 4; nt++)
        #pragma unroll
        for (int r = 0; r < 8; r++) {
            int row = rbase + sub*16 + r, col = n0 + nt*16 + nlane;
            Cf[(long)row*N + col] = acc[sub*4+nt][r] + Res[(long)row*N + col];
        }
    } else {
        #pragma unroll
        for (int sub = 0; sub < 2; sub++)
        #pragma unroll
        for (int nt = 0; nt < 4; nt++)
        #pragma unroll
        for (int r = 0; r < 8; r++) {
            int row = rbase + sub*16 + r, col = n0 + nt*16 + nlane;
            float v = acc[sub*4+nt][r];
            Cb[(long)e*sC + (long)row*N + col] = f2bf(v > 0.f ? v : 0.f);
        }
    }
}

// -------- RoPE + pad to HDP, bf16; Q/K token-major, V FEATURE-major --------
__global__ void rope_pad_kernel(const float* __restrict__ qkv,
                                unsigned short* qpad, unsigned short* kpad, unsigned short* vT)
{
    int rowflat = blockIdx.x;              // b*T + t
    int b = rowflat / T_SEQ, t = rowflat % T_SEQ;
    int tid = threadIdx.x;
    const float* qr = qkv + (long)rowflat * 576;
    if (tid < 288) {                       // data pairs (576 cols)
        int col = tid * 2;
        float xe = qr[col], xo = qr[col+1];
        if (col < 480) {                   // q or k -> rope
            unsigned short* dst; int d;
            if (col < 384) { int head = col / HDn; d = col % HDn;
                dst = qpad + (((long)(b*HQn + head))*T_SEQ + t)*HDP; }
            else { int cc = col - 384; int head = cc / HDn; d = cc % HDn;
                dst = kpad + (((long)(b*HKVn + head))*T_SEQ + t)*HDP; }
            int i = d >> 1;
            float freq = __powf(10000.0f, -(float)(2*i)/(float)HDn);
            float ang = (float)t * freq;
            float cs = __cosf(ang), sn = __sinf(ang);
            dst[d]   = f2bf(xe*cs - xo*sn);
            dst[d+1] = f2bf(xe*sn + xo*cs);
        } else {                           // v -> feature-major [bkv][d][t]
            int cc = col - 480; int head = cc / HDn; int d = cc % HDn;
            long base = ((long)(b*HKVn + head))*HDP*T_SEQ;
            vT[base + (long)d*T_SEQ + t]     = f2bf(xe);
            vT[base + (long)(d+1)*T_SEQ + t] = f2bf(xo);
        }
    } else {                               // zero pad region dims 48..63
        int pid  = tid - 288;              // 0..95
        int unit = pid >> 3, pp = pid & 7;
        int d = HDn + pp*2;
        if (unit < HQn) {
            unsigned short* dst = qpad + (((long)(b*HQn + unit))*T_SEQ + t)*HDP;
            dst[d] = 0; dst[d+1] = 0;
        } else if (unit < HQn + HKVn) {
            unsigned short* dst = kpad + (((long)(b*HKVn + unit-HQn))*T_SEQ + t)*HDP;
            dst[d] = 0; dst[d+1] = 0;
        } else {
            long base = ((long)(b*HKVn + unit-HQn-HKVn))*HDP*T_SEQ;
            vT[base + (long)d*T_SEQ + t]     = 0;
            vT[base + (long)(d+1)*T_SEQ + t] = 0;
        }
    }
}

// ---------------- flash attention (causal GQA), WMMA bf16 ----------------
// grid (T/128, B*HQ), block 256 = 8 waves, wave = 16 q-rows
__global__ __launch_bounds__(256) void attn_kernel(
        const unsigned short* __restrict__ qpad, const unsigned short* __restrict__ kpad,
        const unsigned short* __restrict__ vT, unsigned short* __restrict__ obf)
{
    __shared__ unsigned int ldsK[1024];    // 32 keys x 32 dwords (64 feats, key-major)
    __shared__ unsigned int ldsVt[1024];   // 64 feats x 16 dwords (32 keys, feat-major)
    __shared__ unsigned int ldsP[2048];    // 8 waves x 16 rows x 16 dwords (P tile)
    int tid = threadIdx.x, lane = tid & 31, w = tid >> 5;
    int qblk = blockIdx.x, bh = blockIdx.y;
    int b = bh / HQn, h = bh % HQn, hk = h / (HQn/HKVn);
    int bkv = b*HKVn + hk;
    const unsigned int* qg = (const unsigned int*)qpad + ((long)bh*T_SEQ)*32;
    const unsigned short* kg16 = kpad + ((long)bkv*T_SEQ)*HDP;
    const unsigned short* vg16 = vT + (long)bkv*HDP*T_SEQ;
    const unsigned int* kg = (const unsigned int*)kg16;
    const unsigned int* vg = (const unsigned int*)vg16;
    int q0 = qblk*128 + w*16;
    int koff2 = (lane < 16) ? 0 : 4;
    int jb    = (lane < 16) ? 0 : 8;
    int nlane = lane & 15;
    int halfo = (lane < 16) ? 0 : 8;
    unsigned short* ldsP16 = (unsigned short*)ldsP;

    FragA a0, a1;                          // Q fragments K=0..31 and 32..63
    const unsigned int* qr = qg + (long)(q0 + nlane)*32;
    #pragma unroll
    for (int j = 0; j < 4; j++) {
        a0.u[j] = qr[koff2+j];    a0.u[4+j] = qr[8+koff2+j];
        a1.u[j] = qr[16+koff2+j]; a1.u[4+j] = qr[24+koff2+j];
    }
    float mrow[8], lrow[8];
    v8f o0 = {}, o1 = {}, o2a = {}, o3 = {};
    #pragma unroll
    for (int r = 0; r < 8; r++) { mrow[r] = -1e30f; lrow[r] = 0.f; }

    int nkt = qblk*4 + 4;                  // key tiles of 32 up to block's causal max
    for (int kt = 0; kt < nkt; kt++) {
        int k0 = kt * 32;
#if HAVE_ASYNC
        {                                  // K tile: 32 keys x 128B -> 256 b128
            int key = tid >> 3, seg = tid & 7;
            async_b128(kg16 + (long)(k0 + key)*HDP + seg*8,
                       (char*)ldsK + key*128 + seg*16);
        }
        {                                  // V tile: 64 feats x 64B -> 256 b128
            int f = tid >> 2, kc = (tid & 3)*8;
            async_b128(vg16 + (long)f*T_SEQ + k0 + kc,
                       (char*)ldsVt + f*64 + kc*2);
        }
        async_wait0();
#else
        for (int s = tid; s < 1024; s += 256)
            ldsK[s] = kg[(long)(k0 + (s >> 5))*32 + (s & 31)];
        for (int s = tid; s < 1024; s += 256)
            ldsVt[s] = vg[(long)(s >> 4)*(T_SEQ/2) + (k0 >> 1) + (s & 15)];
#endif
        __syncthreads();

        v8f s0 = {}, s1 = {};              // S tile 16 x 32 (2 x 16x16)
        {
            FragA bk;
            #pragma unroll
            for (int j = 0; j < 8; j++) bk.u[j] = ldsK[(nlane)*32 + koff2 + (j & 3) + ((j >> 2) << 3)];
            s0 = WMMA_BF16(a0, bk, s0);
            #pragma unroll
            for (int j = 0; j < 8; j++) bk.u[j] = ldsK[(nlane)*32 + 16 + koff2 + (j & 3) + ((j >> 2) << 3)];
            s0 = WMMA_BF16(a1, bk, s0);
            #pragma unroll
            for (int j = 0; j < 8; j++) bk.u[j] = ldsK[(16 + nlane)*32 + koff2 + (j & 3) + ((j >> 2) << 3)];
            s1 = WMMA_BF16(a0, bk, s1);
            #pragma unroll
            for (int j = 0; j < 8; j++) bk.u[j] = ldsK[(16 + nlane)*32 + 16 + koff2 + (j & 3) + ((j >> 2) << 3)];
            s1 = WMMA_BF16(a1, bk, s1);
        }
        // scale + causal mask + online softmax (row in C-layout: VGPR r, lane halves)
        #pragma unroll
        for (int r = 0; r < 8; r++) {
            int rowt = q0 + r + halfo;
            float v0 = s0[r] * 0.14433756729740645f;
            float v1 = s1[r] * 0.14433756729740645f;
            if (k0 + nlane      > rowt) v0 = -1e30f;
            if (k0 + 16 + nlane > rowt) v1 = -1e30f;
            float mx = fmaxf(v0, v1);
            #pragma unroll
            for (int off = 1; off < 16; off <<= 1) mx = fmaxf(mx, __shfl_xor(mx, off, 32));
            float mnew = fmaxf(mrow[r], mx);
            float p0 = __expf(v0 - mnew);
            float p1 = __expf(v1 - mnew);
            float ps = p0 + p1;
            #pragma unroll
            for (int off = 1; off < 16; off <<= 1) ps += __shfl_xor(ps, off, 32);
            float alpha = __expf(mrow[r] - mnew);
            mrow[r] = mnew;
            lrow[r] = lrow[r]*alpha + ps;
            o0[r] *= alpha; o1[r] *= alpha; o2a[r] *= alpha; o3[r] *= alpha;
            int rm = r + halfo;
            ldsP16[(w << 9) + rm*32 + nlane]      = f2bf(p0);
            ldsP16[(w << 9) + rm*32 + 16 + nlane] = f2bf(p1);
        }
        __syncthreads();
        FragA pf;                           // P as A-fragment (16x32)
        const unsigned int* pr = &ldsP[(w << 8) + nlane*16];
        #pragma unroll
        for (int j = 0; j < 4; j++) { pf.u[j] = pr[koff2+j]; pf.u[4+j] = pr[8+koff2+j]; }
        FragA bv;
        #pragma unroll
        for (int j = 0; j < 8; j++) bv.u[j] = ldsVt[(nlane)*16 + koff2 + (j & 3) + ((j >> 2) << 3)];
        o0  = WMMA_BF16(pf, bv, o0);
        #pragma unroll
        for (int j = 0; j < 8; j++) bv.u[j] = ldsVt[(16 + nlane)*16 + koff2 + (j & 3) + ((j >> 2) << 3)];
        o1  = WMMA_BF16(pf, bv, o1);
        #pragma unroll
        for (int j = 0; j < 8; j++) bv.u[j] = ldsVt[(32 + nlane)*16 + koff2 + (j & 3) + ((j >> 2) << 3)];
        o2a = WMMA_BF16(pf, bv, o2a);
        #pragma unroll
        for (int j = 0; j < 8; j++) bv.u[j] = ldsVt[(48 + nlane)*16 + koff2 + (j & 3) + ((j >> 2) << 3)];
        o3  = WMMA_BF16(pf, bv, o3);
        __syncthreads();
    }
    #pragma unroll
    for (int r = 0; r < 8; r++) {           // store first 48 feats only
        int tpos = q0 + r + halfo;
        long ob = ((long)b*T_SEQ + tpos)*C_DIM + h*HDn;
        float inv = 1.0f / lrow[r];
        obf[ob + nlane]      = f2bf(o0[r]  * inv);
        obf[ob + 16 + nlane] = f2bf(o1[r]  * inv);
        obf[ob + 32 + nlane] = f2bf(o2a[r] * inv);
    }
}

// ---------------- MoE router: logits + softmax + top1 (wave per row) ----------------
__global__ void router_kernel(const float* __restrict__ flat, const float* __restrict__ Wr,
                              const float* __restrict__ br, float* probs, float* top1v,
                              int* top1i, float* sum_probs)
{
    int tid = threadIdx.x, lane = tid & 31, wid = tid >> 5;
    int row = blockIdx.x*8 + wid;
    const float* fr = flat + (long)row * C_DIM;
    float acc[E_NUM] = {0.f, 0.f, 0.f, 0.f};
    for (int c = lane; c < C_DIM; c += 32) {
        float xv = fr[c];
        #pragma unroll
        for (int e = 0; e < E_NUM; e++) acc[e] += xv * Wr[c*E_NUM + e];
    }
    #pragma unroll
    for (int e = 0; e < E_NUM; e++)
        for (int off = 16; off; off >>= 1) acc[e] += __shfl_xor(acc[e], off, 32);
    if (lane == 0) {
        float mx = -1e30f;
        #pragma unroll
        for (int e = 0; e < E_NUM; e++) { acc[e] += br[e]; mx = fmaxf(mx, acc[e]); }
        float s = 0.f;
        #pragma unroll
        for (int e = 0; e < E_NUM; e++) { acc[e] = __expf(acc[e]-mx); s += acc[e]; }
        float inv = 1.0f/s; int bi = 0; float bv = -1.f;
        #pragma unroll
        for (int e = 0; e < E_NUM; e++) {
            float p = acc[e]*inv;
            probs[(long)row*E_NUM + e] = p;
            atomicAdd(&sum_probs[e], p);
            if (p > bv) { bv = p; bi = e; }
        }
        top1v[row] = bv; top1i[row] = bi;
    }
}

// ---------------- order-dependent capacity scan (single 1024-thread block) ----------------
__global__ __launch_bounds__(1024) void scan_kernel(const int* __restrict__ top1i,
                                                    int* slot, float* sums /* [4..7]=disp */)
{
    __shared__ int sc[E_NUM*1024];
    int t = threadIdx.x;
    int cnt[E_NUM] = {0,0,0,0};
    int eo[8];
    #pragma unroll
    for (int i = 0; i < 8; i++) { int e = top1i[t*8+i]; eo[i] = e; cnt[e]++; }
    #pragma unroll
    for (int e = 0; e < E_NUM; e++) sc[e*1024 + t] = cnt[e];
    __syncthreads();
    for (int off = 1; off < 1024; off <<= 1) {
        int v[E_NUM];
        #pragma unroll
        for (int e = 0; e < E_NUM; e++) v[e] = (t >= off) ? sc[e*1024 + t - off] : 0;
        __syncthreads();
        #pragma unroll
        for (int e = 0; e < E_NUM; e++) sc[e*1024 + t] += v[e];
        __syncthreads();
    }
    int run[E_NUM];
    #pragma unroll
    for (int e = 0; e < E_NUM; e++) run[e] = sc[e*1024 + t] - cnt[e];   // exclusive
    #pragma unroll
    for (int i = 0; i < 8; i++) {
        int e = eo[i]; int pos = run[e]++;
        slot[t*8+i] = (pos < CAPS) ? pos : -1;
    }
    if (t == 0) {
        #pragma unroll
        for (int e = 0; e < E_NUM; e++) {
            int tot = sc[e*1024 + 1023];
            sums[4+e] = (float)(tot < CAPS ? tot : CAPS);
        }
    }
}

// ---------------- scatter to expert buffers (bf16) ----------------
__global__ void scatter_kernel(const unsigned short* __restrict__ flatbf,
                               const int* __restrict__ top1i, const int* __restrict__ slot,
                               unsigned short* __restrict__ buf)
{
    int row = blockIdx.x, tid = threadIdx.x;
    int s = slot[row];
    if (s < 0) return;
    int e = top1i[row];
    const unsigned short* src = flatbf + (long)row*C_DIM;
    unsigned short* dst = buf + ((long)e*CAPS + s)*C_DIM;
    #pragma unroll
    for (int i = 0; i < 3; i++) dst[tid + 128*i] = src[tid + 128*i];
}

// ---------------- gather + gate + residual + aux ----------------
__global__ void gather_kernel(const float* __restrict__ x1, const float* __restrict__ o2,
                              const float* __restrict__ top1v, const int* __restrict__ top1i,
                              const int* __restrict__ slot, const float* __restrict__ sums,
                              float* __restrict__ out)
{
    int row = blockIdx.x, tid = threadIdx.x;
    int s = slot[row];
    float g = 0.f; long ob = 0;
    if (s >= 0) { g = top1v[row]; ob = ((long)top1i[row]*CAPS + s)*C_DIM; }
    #pragma unroll
    for (int i = 0; i < 3; i++) {
        int c = tid + 128*i;
        float v = x1[(long)row*C_DIM + c];
        if (s >= 0) v += o2[ob + c] * g;
        out[(long)row*C_DIM + c] = v;
    }
    if (row == 0 && tid == 0) {
        float aux = 0.f;
        #pragma unroll
        for (int e = 0; e < E_NUM; e++)
            aux += (sums[e]/(float)NROWS) * (sums[4+e]/(float)NROWS);
        out[(long)NROWS*C_DIM] = aux * (float)E_NUM;
    }
}

// ======================= host launcher =======================
extern "C" void kernel_launch(void* const* d_in, const int* in_sizes, int n_in,
                              void* d_out, int out_size, void* d_ws, size_t ws_size,
                              hipStream_t stream)
{
    (void)in_sizes; (void)n_in; (void)out_size; (void)ws_size;
    const float* x    = (const float*)d_in[0];
    const int*   aid  = (const int*)  d_in[1];
    const float* ln1g = (const float*)d_in[2];
    const float* ln1b = (const float*)d_in[3];
    const float* ln2g = (const float*)d_in[4];
    const float* ln2b = (const float*)d_in[5];
    const float* Wq   = (const float*)d_in[6];
    const float* Aq   = (const float*)d_in[7];
    const float* Bq   = (const float*)d_in[8];
    const float* Wk   = (const float*)d_in[9];
    const float* Ak   = (const float*)d_in[10];
    const float* Bk   = (const float*)d_in[11];
    const float* Wv   = (const float*)d_in[12];
    const float* Av   = (const float*)d_in[13];
    const float* Bv   = (const float*)d_in[14];
    const float* Wo   = (const float*)d_in[15];
    const float* Ao   = (const float*)d_in[16];
    const float* Bo   = (const float*)d_in[17];
    const float* Wr   = (const float*)d_in[18];
    const float* br   = (const float*)d_in[19];
    const float* W1   = (const float*)d_in[20];
    const float* A1   = (const float*)d_in[21];
    const float* B1   = (const float*)d_in[22];
    const float* W2   = (const float*)d_in[23];
    const float* A2   = (const float*)d_in[24];
    const float* B2   = (const float*)d_in[25];
    float* out = (float*)d_out;

    char* wp = (char*)d_ws;
    auto alloc = [&](size_t bytes) -> char* {
        char* p = wp; wp += (bytes + 255) & ~(size_t)255; return p;
    };
    unsigned short* h_bf   = (unsigned short*)alloc((size_t)NROWS*C_DIM*2);
    unsigned short* Wqkv_e = (unsigned short*)alloc((size_t)C_DIM*576*2);       // [576,384]
    unsigned short* Wo_e   = (unsigned short*)alloc((size_t)C_DIM*C_DIM*2);     // [384,384]
    unsigned short* W1_e   = (unsigned short*)alloc((size_t)E_NUM*C_DIM*FFN*2); // [e][1536,384]
    unsigned short* W2_e   = (unsigned short*)alloc((size_t)E_NUM*FFN*C_DIM*2); // [e][384,1536]
    float*          qkv    = (float*)         alloc((size_t)NROWS*576*4);
    unsigned short* qpad   = (unsigned short*)alloc((size_t)B_NUM*HQn*T_SEQ*HDP*2);
    unsigned short* kpad   = (unsigned short*)alloc((size_t)B_NUM*HKVn*T_SEQ*HDP*2);
    unsigned short* vTm    = (unsigned short*)alloc((size_t)B_NUM*HKVn*HDP*T_SEQ*2);
    unsigned short* obf    = (unsigned short*)alloc((size_t)NROWS*C_DIM*2);
    float*          x1     = (float*)         alloc((size_t)NROWS*C_DIM*4);
    float*          flatf  = (float*)         alloc((size_t)NROWS*C_DIM*4);
    unsigned short* flatbf = (unsigned short*)alloc((size_t)NROWS*C_DIM*2);
    float*          probs  = (float*)         alloc((size_t)NROWS*E_NUM*4);
    float*          top1v  = (float*)         alloc((size_t)NROWS*4);
    int*            top1i  = (int*)           alloc((size_t)NROWS*4);
    int*            slot   = (int*)           alloc((size_t)NROWS*4);
    unsigned short* buf    = (unsigned short*)alloc((size_t)E_NUM*CAPS*C_DIM*2 + 32);
    float*          sums   = (float*)(buf + (size_t)E_NUM*CAPS*C_DIM); // [0..3]=probs [4..7]=disp
    unsigned short* h1     = (unsigned short*)alloc((size_t)E_NUM*CAPS*FFN*2);
    float*          o2m    = (float*)         alloc((size_t)E_NUM*CAPS*C_DIM*4);

    // 1) zero expert buffers + running sums (contiguous span)
    long zero_dw = ((long)E_NUM*CAPS*C_DIM*2)/4 + 8;
    zero_u32<<<2048, 256, 0, stream>>>((unsigned int*)buf, zero_dw);

    // 2) fold LoRA into effective transposed bf16 weights
    prep_weight<<<dim3(576, 1, 1), 256, 0, stream>>>(Wq, 0, Aq, 0, (long)R_LORA*C_DIM,
        Bq, 0, (long)384*R_LORA, Wqkv_e, 0, C_DIM, 384, 0, aid);
    prep_weight<<<dim3(144, 1, 1), 256, 0, stream>>>(Wk, 0, Ak, 0, (long)R_LORA*C_DIM,
        Bk, 0, (long)96*R_LORA, Wqkv_e, 0, C_DIM, 96, 384, aid);
    prep_weight<<<dim3(144, 1, 1), 256, 0, stream>>>(Wv, 0, Av, 0, (long)R_LORA*C_DIM,
        Bv, 0, (long)96*R_LORA, Wqkv_e, 0, C_DIM, 96, 480, aid);
    prep_weight<<<dim3(576, 1, 1), 256, 0, stream>>>(Wo, 0, Ao, 0, (long)R_LORA*C_DIM,
        Bo, 0, (long)C_DIM*R_LORA, Wo_e, 0, C_DIM, C_DIM, 0, aid);
    prep_weight<<<dim3(1024, 1, E_NUM), 256, 0, stream>>>(W1, (long)C_DIM*FFN,
        A1, (long)4*R_LORA*C_DIM, (long)R_LORA*C_DIM,
        B1, (long)4*FFN*R_LORA,   (long)FFN*R_LORA,
        W1_e, (long)C_DIM*FFN, C_DIM, FFN, 0, aid);
    prep_weight<<<dim3(1024, 1, E_NUM), 256, 0, stream>>>(W2, (long)FFN*C_DIM,
        A2, (long)4*R_LORA*FFN,   (long)R_LORA*FFN,
        B2, (long)4*C_DIM*R_LORA, (long)C_DIM*R_LORA,
        W2_e, (long)FFN*C_DIM, FFN, C_DIM, 0, aid);

    // 3) LN1 -> bf16 activations
    ln_kernel<<<NROWS, 128, 0, stream>>>(x, ln1g, ln1b, nullptr, h_bf);

    // 4) fused QKV projection (WMMA)
    gemm_bf16<<<dim3(576/64, NROWS/256, 1), 256, 0, stream>>>(
        h_bf, 0, Wqkv_e, 0, qkv, nullptr, 0, nullptr, NROWS, 576, C_DIM, 0);

    // 5) RoPE + pad/convert (V stored feature-major)
    rope_pad_kernel<<<NROWS, 384, 0, stream>>>(qkv, qpad, kpad, vTm);

    // 6) causal GQA flash attention (WMMA)
    attn_kernel<<<dim3(T_SEQ/128, B_NUM*HQn), 256, 0, stream>>>(qpad, kpad, vTm, obf);

    // 7) O projection + residual (WMMA)
    gemm_bf16<<<dim3(C_DIM/64, NROWS/256, 1), 256, 0, stream>>>(
        obf, 0, Wo_e, 0, x1, nullptr, 0, x, NROWS, C_DIM, C_DIM, 1);

    // 8) LN2
    ln_kernel<<<NROWS, 128, 0, stream>>>(x1, ln2g, ln2b, flatf, flatbf);

    // 9) router + top1
    router_kernel<<<NROWS/8, 256, 0, stream>>>(flatf, Wr, br, probs, top1v, top1i, sums);

    // 10) capacity scan
    scan_kernel<<<1, 1024, 0, stream>>>(top1i, slot, sums);

    // 11) scatter tokens to expert buffers
    scatter_kernel<<<NROWS, 128, 0, stream>>>(flatbf, top1i, slot, buf);

    // 12) expert FFN layer 1 (WMMA, relu -> bf16)
    gemm_bf16<<<dim3(FFN/64, CAPS/256, E_NUM), 256, 0, stream>>>(
        buf, (long)CAPS*C_DIM, W1_e, (long)C_DIM*FFN,
        nullptr, h1, (long)CAPS*FFN, nullptr, CAPS, FFN, C_DIM, 2);

    // 13) expert FFN layer 2 (WMMA)
    gemm_bf16<<<dim3(C_DIM/64, CAPS/256, E_NUM), 256, 0, stream>>>(
        h1, (long)CAPS*FFN, W2_e, (long)FFN*C_DIM,
        o2m, nullptr, (long)CAPS*C_DIM, nullptr, CAPS, C_DIM, FFN, 0);

    // 14) gather + gate + residual + aux
    gather_kernel<<<NROWS, 128, 0, stream>>>(x1, o2m, top1v, top1i, slot, sums, out);
}
